// CausalAttention_53077205844052
// MI455X (gfx1250) — matrix operand order repaired and checked by reading, not compiled, not run on hardware
//
#include <hip/hip_runtime.h>

#define B_    4
#define S_    4096
#define DIN_  1024
#define DOUT_ 64

typedef __attribute__((ext_vector_type(16))) __bf16 v16bf;
typedef __attribute__((ext_vector_type(2)))  __bf16 v2bf;
typedef __attribute__((ext_vector_type(8)))  float  v8f;

union Frag {
    v16bf v;
    unsigned u[8];
    uint4 q[2];
};

__device__ __forceinline__ unsigned packcvt(float lo, float hi) {
    v2bf t;
    t[0] = (__bf16)lo;
    t[1] = (__bf16)hi;
    return __builtin_bit_cast(unsigned, t);
}
__device__ __forceinline__ unsigned short cvt1(float f) {
    __bf16 h = (__bf16)f;
    return __builtin_bit_cast(unsigned short, h);
}

// ---------------------------------------------------------------------------
// Kernel 0: bulk fp32 -> bf16 conversion (bandwidth-bound, packed cvt).
// ---------------------------------------------------------------------------
__global__ __launch_bounds__(256) void to_bf16(
    const float* __restrict__ src, unsigned short* __restrict__ dst, int n)
{
    const int i = (int)(blockIdx.x * blockDim.x + threadIdx.x) * 8;
    if (i >= n) return;
    const float4 a = *(const float4*)(src + i);
    const float4 b = *(const float4*)(src + i + 4);
    uint4 o;
    o.x = packcvt(a.x, a.y);
    o.y = packcvt(a.z, a.w);
    o.z = packcvt(b.x, b.y);
    o.w = packcvt(b.z, b.w);
    *(uint4*)(dst + i) = o;
}

// ---------------------------------------------------------------------------
// Kernel 1: QKV projection from bf16 inputs.  One wave = 16x64 output tile.
// Q, K stored row-major bf16 [B,S,64]; V stored transposed bf16 [B,64,S].
// All fragment loads are contiguous 16B (global_load_b128).
// ---------------------------------------------------------------------------
__global__ __launch_bounds__(256) void qkv_proj(
    const unsigned short* __restrict__ xb,
    const unsigned short* __restrict__ Wqb,
    const unsigned short* __restrict__ Wkb,
    const unsigned short* __restrict__ Wvb,
    unsigned short* __restrict__ Qb,
    unsigned short* __restrict__ Kb,
    unsigned short* __restrict__ Vt)
{
    const int wid  = (int)((blockIdx.x * blockDim.x + threadIdx.x) >> 5);
    const int lane = (int)(threadIdx.x & 31);
    const int r    = lane & 15;
    const int hf   = lane >> 4;
    if (wid >= B_ * (S_ / 16) * 3) return;

    const int mat  = wid % 3;
    const int tile = wid / 3;
    const int b    = tile >> 8;          // tile / (S/16)
    const int m0   = (tile & 255) << 4;  // 16-row tile base

    const unsigned short* W = (mat == 0) ? Wqb : (mat == 1) ? Wkb : Wvb;

    v8f acc[4];
#pragma unroll
    for (int t = 0; t < 4; ++t)
#pragma unroll
        for (int i = 0; i < 8; ++i) acc[t][i] = 0.0f;

    const unsigned short* xrow = xb + ((size_t)b * S_ + m0 + r) * DIN_;

    for (int kk = 0; kk < DIN_; kk += 32) {
        // A fragment: v0..3 = K[8*hf .. 8*hf+7], v4..7 = K[16+8*hf ..]
        Frag a;
        a.q[0] = *(const uint4*)(xrow + kk + 8 * hf);
        a.q[1] = *(const uint4*)(xrow + kk + 16 + 8 * hf);
#pragma unroll
        for (int t = 0; t < 4; ++t) {
            // B fragment: W^T chunk; W row contiguous along K.
            const unsigned short* wc = W + (size_t)(t * 16 + r) * DIN_ + kk;
            Frag bb;
            bb.q[0] = *(const uint4*)(wc + 8 * hf);
            bb.q[1] = *(const uint4*)(wc + 16 + 8 * hf);
            acc[t] = __builtin_amdgcn_wmma_f32_16x16x32_bf16(
                false, a.v, false, bb.v, (short)0, acc[t], false, false);
        }
    }

    if (mat < 2) {
        unsigned short* out = (mat == 0 ? Qb : Kb) + ((size_t)b * S_ + m0) * DOUT_;
#pragma unroll
        for (int t = 0; t < 4; ++t)
#pragma unroll
            for (int i = 0; i < 8; ++i)
                out[(size_t)(i + 8 * hf) * DOUT_ + t * 16 + r] = cvt1(acc[t][i]);
    } else {
        unsigned short* out = Vt + (size_t)b * DOUT_ * S_;
#pragma unroll
        for (int t = 0; t < 4; ++t)
#pragma unroll
            for (int i = 0; i < 8; ++i)
                out[(size_t)(t * 16 + r) * S_ + m0 + i + 8 * hf] = cvt1(acc[t][i]);
    }
}

// ---------------------------------------------------------------------------
// Kernel 2: flash causal attention.  One wave owns 16 query rows; streams
// 32-key blocks with online softmax; 8 bf16 WMMAs per block.
// ---------------------------------------------------------------------------
__global__ __launch_bounds__(256) void attn(
    const unsigned short* __restrict__ Qb,
    const unsigned short* __restrict__ Kb,
    const unsigned short* __restrict__ Vt,
    float* __restrict__ out)
{
    __shared__ __align__(16) unsigned short smem[8 * 16 * 32]; // 1KB/wave P tile

    const int winb = (int)(threadIdx.x >> 5);
    const int lane = (int)(threadIdx.x & 31);
    const int r    = lane & 15;
    const int hf   = lane >> 4;
    const int wid  = (int)blockIdx.x * ((int)blockDim.x >> 5) + winb;
    if (wid >= B_ * (S_ / 16)) return;

    const int b  = wid >> 8;           // wid / (S/16)
    const int q0 = (wid & 255) << 4;

    unsigned short* myLds = smem + winb * (16 * 32);

    // Load Q A-fragments once: 2 K-chunks of 32 dims, all b128 loads.
    Frag qa[2];
    {
        const unsigned short* qrow = Qb + ((size_t)b * S_ + q0 + r) * DOUT_;
#pragma unroll
        for (int c = 0; c < 2; ++c) {
            qa[c].q[0] = *(const uint4*)(qrow + c * 32 + 8 * hf);
            qa[c].q[1] = *(const uint4*)(qrow + c * 32 + 16 + 8 * hf);
        }
    }

    v8f o[4];
#pragma unroll
    for (int t = 0; t < 4; ++t)
#pragma unroll
        for (int i = 0; i < 8; ++i) o[t][i] = 0.0f;

    float M[8], L[8];
#pragma unroll
    for (int i = 0; i < 8; ++i) { M[i] = -__builtin_inff(); L[i] = 0.0f; }

    const int nb = (q0 >> 5) + 1;      // number of 32-key blocks (causal)
    for (int j = 0; j < nb * 32; j += 32) {
        // ---- scores: S = Q x K^T  (two 16-key tiles, two 32-dim chunks) ----
        v8f s0, s1;
#pragma unroll
        for (int i = 0; i < 8; ++i) { s0[i] = 0.0f; s1[i] = 0.0f; }
        const unsigned short* kr0 = Kb + ((size_t)b * S_ + j + r) * DOUT_;
        const unsigned short* kr1 = Kb + ((size_t)b * S_ + j + 16 + r) * DOUT_;
#pragma unroll
        for (int c = 0; c < 2; ++c) {
            Frag kf0, kf1;
            kf0.q[0] = *(const uint4*)(kr0 + c * 32 + 8 * hf);
            kf0.q[1] = *(const uint4*)(kr0 + c * 32 + 16 + 8 * hf);
            kf1.q[0] = *(const uint4*)(kr1 + c * 32 + 8 * hf);
            kf1.q[1] = *(const uint4*)(kr1 + c * 32 + 16 + 8 * hf);
            s0 = __builtin_amdgcn_wmma_f32_16x16x32_bf16(
                false, qa[c].v, false, kf0.v, (short)0, s0, false, false);
            s1 = __builtin_amdgcn_wmma_f32_16x16x32_bf16(
                false, qa[c].v, false, kf1.v, (short)0, s1, false, false);
        }

        // ---- scale + causal mask ----
#pragma unroll
        for (int i = 0; i < 8; ++i) { s0[i] *= 0.125f; s1[i] *= 0.125f; }
        if (j + 31 > q0) {
            const int qbase = q0 + 8 * hf;
#pragma unroll
            for (int i = 0; i < 8; ++i) {
                if (j + r      > qbase + i) s0[i] = -__builtin_inff();
                if (j + 16 + r > qbase + i) s1[i] = -__builtin_inff();
            }
        }

        // ---- online softmax (row stats live across 16-lane halves) ----
        float nm[8], f[8], p0[8], p1[8];
#pragma unroll
        for (int i = 0; i < 8; ++i) {
            float t = fmaxf(s0[i], s1[i]);
            t = fmaxf(t, __shfl_xor(t, 1));
            t = fmaxf(t, __shfl_xor(t, 2));
            t = fmaxf(t, __shfl_xor(t, 4));
            t = fmaxf(t, __shfl_xor(t, 8));
            nm[i] = fmaxf(M[i], t);
            f[i]  = __expf(M[i] - nm[i]);
            p0[i] = __expf(s0[i] - nm[i]);
            p1[i] = __expf(s1[i] - nm[i]);
            float rs = p0[i] + p1[i];
            rs += __shfl_xor(rs, 1);
            rs += __shfl_xor(rs, 2);
            rs += __shfl_xor(rs, 4);
            rs += __shfl_xor(rs, 8);
            L[i] = L[i] * f[i] + rs;
            M[i] = nm[i];
        }
#pragma unroll
        for (int t = 0; t < 4; ++t)
#pragma unroll
            for (int i = 0; i < 8; ++i) o[t][i] *= f[i];

        // ---- re-layout P (C-format -> A-format) through LDS ----
#pragma unroll
        for (int i = 0; i < 8; ++i) {
            myLds[(i + 8 * hf) * 32 + r]      = cvt1(p0[i]);
            myLds[(i + 8 * hf) * 32 + 16 + r] = cvt1(p1[i]);
        }
        asm volatile("s_wait_dscnt 0" ::: "memory");   // same-wave DS RAW

        Frag pa;
        {
            const unsigned short* lrow = myLds + r * 32;
            pa.q[0] = *(const uint4*)(lrow + 8 * hf);
            pa.q[1] = *(const uint4*)(lrow + 16 + 8 * hf);
        }

        // ---- O += P x V  (V transposed -> contiguous b128 loads) ----
#pragma unroll
        for (int t = 0; t < 4; ++t) {
            const unsigned short* vrow =
                Vt + ((size_t)b * DOUT_ + t * 16 + r) * S_ + j;
            Frag vb;
            vb.q[0] = *(const uint4*)(vrow + 8 * hf);
            vb.q[1] = *(const uint4*)(vrow + 16 + 8 * hf);
            o[t] = __builtin_amdgcn_wmma_f32_16x16x32_bf16(
                false, pa.v, false, vb.v, (short)0, o[t], false, false);
        }
    }

    // ---- normalize and write fp32 output ----
    float invL[8];
#pragma unroll
    for (int i = 0; i < 8; ++i) invL[i] = 1.0f / L[i];
#pragma unroll
    for (int t = 0; t < 4; ++t)
#pragma unroll
        for (int i = 0; i < 8; ++i)
            out[((size_t)b * S_ + q0 + i + 8 * hf) * DOUT_ + t * 16 + r] =
                o[t][i] * invL[i];
}

// ---------------------------------------------------------------------------
extern "C" void kernel_launch(void* const* d_in, const int* in_sizes, int n_in,
                              void* d_out, int out_size, void* d_ws, size_t ws_size,
                              hipStream_t stream) {
    const float* x  = (const float*)d_in[0];
    const float* Wq = (const float*)d_in[1];
    const float* Wk = (const float*)d_in[2];
    const float* Wv = (const float*)d_in[3];
    float* outp = (float*)d_out;

    const size_t nX = (size_t)B_ * S_ * DIN_;    // 16,777,216
    const size_t nW = (size_t)DOUT_ * DIN_;      // 65,536
    const size_t nQ = (size_t)B_ * S_ * DOUT_;   // 1,048,576

    unsigned short* xb  = (unsigned short*)d_ws;         // [B,S,1024] bf16
    unsigned short* Wqb = xb  + nX;                      // [64,1024]  bf16
    unsigned short* Wkb = Wqb + nW;
    unsigned short* Wvb = Wkb + nW;
    unsigned short* Qb  = Wvb + nW;                      // [B,S,64]   bf16
    unsigned short* Kb  = Qb  + nQ;                      // [B,S,64]   bf16
    unsigned short* Vt  = Kb  + nQ;                      // [B,64,S]   bf16

    // Kernel 0: fp32 -> bf16 conversions (8 elements/thread)
    to_bf16<<<(int)(nX / 8 / 256), 256, 0, stream>>>(x,  xb,  (int)nX);
    to_bf16<<<(int)((nW / 8 + 255) / 256), 256, 0, stream>>>(Wq, Wqb, (int)nW);
    to_bf16<<<(int)((nW / 8 + 255) / 256), 256, 0, stream>>>(Wk, Wkb, (int)nW);
    to_bf16<<<(int)((nW / 8 + 255) / 256), 256, 0, stream>>>(Wv, Wvb, (int)nW);

    // Kernel 1: 3072 waves (B * S/16 tiles * 3 matrices), 8 waves/block
    const int w1 = B_ * (S_ / 16) * 3;
    qkv_proj<<<(w1 * 32 + 255) / 256, 256, 0, stream>>>(xb, Wqb, Wkb, Wvb, Qb, Kb, Vt);

    // Kernel 2: 1024 waves (B * S/16 query tiles), 8 waves/block
    const int w2 = B_ * (S_ / 16);
    attn<<<(w2 * 32 + 255) / 256, 256, 0, stream>>>(Qb, Kb, Vt, outp);
}
